// VLAKVPruningAttention_37314675867761
// MI455X (gfx1250) — matrix operand order repaired and checked by reading, not compiled, not run on hardware
//
#include <hip/hip_runtime.h>

// ---------------------------------------------------------------------------
// VLA KV-pruning attention for MI455X (gfx1250, wave32, WMMA bf16,
// async global->LDS staging, double-buffered)
// ---------------------------------------------------------------------------
#define S_LEN       640
#define NHEAD       32
#define HDIM        128
#define HID_DIM     4096
#define VIS_START   1
#define VIS_END     513
#define INSTR_START 513
#define INSTR_END   584          // max(513, 640-56)
#define NVIS        512
#define NKEEP       256
#define NKEPT       384          // 1 + 256 + (640-513)
#define ATTN_SCALE  0.08838834764831845f   // 1/sqrt(128)

#define BLK_M 32                 // block M tile (2 wave sub-tiles)
#define BLK_N 256                // block N tile (8 waves x 32)
#define KSTEPS (HID_DIM / 32)    // 128 k-steps of 32

typedef __attribute__((ext_vector_type(16))) __bf16 v16bf;
typedef __attribute__((ext_vector_type(8)))  float  v8f;

// --- bf16 WMMA wrapper (D = A(16x32) * B(32x16) + C, f32 accum) -------------
static __device__ inline v8f wmma_bf16(v16bf a, v16bf b, v8f c) {
  return __builtin_amdgcn_wmma_f32_16x16x32_bf16(
      /*neg_a=*/false, a, /*neg_b=*/false, b,
      /*c_mod=*/(short)0, c, /*reuse_a=*/false, /*reuse_b=*/false);
}

// B-operand fill: 16 consecutive f32 -> bf16, element i = K = 16*half + i.
static __device__ inline v16bf cvtB16(const float* p) {
  const float4* p4 = (const float4*)p;
  v16bf r;
#pragma unroll
  for (int j = 0; j < 4; ++j) {
    float4 f = p4[j];
    r[4*j+0] = (__bf16)f.x; r[4*j+1] = (__bf16)f.y;
    r[4*j+2] = (__bf16)f.z; r[4*j+3] = (__bf16)f.w;
  }
  return r;
}

// A-operand fill: elems 0..15 = K=koff..koff+15
// (koff = 16*(lane>=16)), matching cvtB16.
static __device__ inline v16bf cvtA16(const float* p, int koff) {
  const float4* p4 = (const float4*)(p + koff);
  float4 f0 = p4[0], f1 = p4[1], f2 = p4[2], f3 = p4[3];
  v16bf r;
  r[0]=(__bf16)f0.x; r[1]=(__bf16)f0.y; r[2]=(__bf16)f0.z; r[3]=(__bf16)f0.w;
  r[4]=(__bf16)f1.x; r[5]=(__bf16)f1.y; r[6]=(__bf16)f1.z; r[7]=(__bf16)f1.w;
  r[8]=(__bf16)f2.x; r[9]=(__bf16)f2.y; r[10]=(__bf16)f2.z; r[11]=(__bf16)f2.w;
  r[12]=(__bf16)f3.x; r[13]=(__bf16)f3.y; r[14]=(__bf16)f3.z; r[15]=(__bf16)f3.w;
  return r;
}

// CDNA5 async global->LDS copy (GVS addressing: SGPR64 base + VGPR32 offset).
// Tracked by ASYNCcnt; completion enforced with s_wait_asynccnt.
static __device__ inline void async_copy_b128(unsigned lds_off, unsigned gbyte_off,
                                              const float* base) {
  asm volatile("global_load_async_to_lds_b128 %0, %1, %2"
               :: "v"(lds_off), "v"(gbyte_off), "s"(base) : "memory");
}

// ---------------------------------------------------------------------------
// Fused QKV / O projection: C[z][640][4096] = A[640][4096] @ W[z][4096][4096]^T
// blockIdx.z selects the weight/output. Block = 8 waves, tile 32(M) x 256(N).
// Wave tile 32x32 = 4 accumulators; A staged via async-LDS, double buffered.
// ---------------------------------------------------------------------------
__global__ __launch_bounds__(256)
void gemm_wmma_kernel(const float* __restrict__ A,
                      const float* __restrict__ W0, const float* __restrict__ W1,
                      const float* __restrict__ W2,
                      float* __restrict__ C0, float* __restrict__ C1,
                      float* __restrict__ C2) {
  __shared__ float lds_a[2][BLK_M][32];     // 8 KB, double-buffered A tile

  const float* W = (blockIdx.z == 0) ? W0 : (blockIdx.z == 1 ? W1 : W2);
  float*       C = (blockIdx.z == 0) ? C0 : (blockIdx.z == 1 ? C1 : C2);

  const int lane = threadIdx.x & 31;
  const int wave = threadIdx.x >> 5;
  const int half = lane >> 4;
  const int l15  = lane & 15;
  const int m0   = blockIdx.y * BLK_M;
  const int n0   = blockIdx.x * BLK_N + wave * 32;

  // Each thread async-copies one b128 of the 32x32 A tile per k-step.
  const int arow_t = threadIdx.x >> 3;      // 0..31
  const int achk_t = threadIdx.x & 7;       // 0..7  (4 floats each)
  const unsigned a_goff0 =
      ((unsigned)(m0 + arow_t) * HID_DIM + achk_t * 4) * 4u;   // bytes
  const unsigned lds_off0 =
      (unsigned)(uintptr_t)&lds_a[0][arow_t][achk_t * 4];      // low 32 = LDS addr
  const unsigned lds_stride = (unsigned)sizeof(float) * BLK_M * 32;

  const float* w0row = W + (size_t)(n0 + l15) * HID_DIM;
  const float* w1row = W + (size_t)(n0 + 16 + l15) * HID_DIM;

  v8f c00 = {}, c01 = {}, c10 = {}, c11 = {};

  // prologue: stage k-step 0 into buffer 0
  async_copy_b128(lds_off0, a_goff0, A);

  for (int ks = 0; ks < KSTEPS; ++ks) {
    const int kt  = ks * 32;
    const int cur = ks & 1;
    if (ks + 1 < KSTEPS) {   // prefetch next k-step into the other buffer
      async_copy_b128(lds_off0 + (cur ^ 1) * lds_stride,
                      a_goff0 + (unsigned)(kt + 32) * 4u, A);
      asm volatile("s_wait_asynccnt 0x1" ::: "memory");  // current buffer done
    } else {
      asm volatile("s_wait_asynccnt 0x0" ::: "memory");
    }
    __syncthreads();                                      // LDS visible to all

    __builtin_prefetch(w0row + kt + 256, 0, 1);           // global_prefetch_b8
    __builtin_prefetch(w1row + kt + 256, 0, 1);

    v16bf a0 = cvtA16(&lds_a[cur][l15][0],      half * 16);
    v16bf a1 = cvtA16(&lds_a[cur][16 + l15][0], half * 16);
    v16bf b0 = cvtB16(w0row + kt + half * 16);
    v16bf b1 = cvtB16(w1row + kt + half * 16);
    c00 = wmma_bf16(a0, b0, c00);
    c01 = wmma_bf16(a0, b1, c01);
    c10 = wmma_bf16(a1, b0, c10);
    c11 = wmma_bf16(a1, b1, c11);

    __syncthreads();   // all waves done reading `cur` before it is overwritten
  }

  const int r0 = m0 + half * 8;
#pragma unroll
  for (int r = 0; r < 8; ++r) {
    C[(size_t)(r0 + r)      * HID_DIM + n0 + l15]      = c00[r];
    C[(size_t)(r0 + r)      * HID_DIM + n0 + 16 + l15] = c01[r];
    C[(size_t)(r0 + 16 + r) * HID_DIM + n0 + l15]      = c10[r];
    C[(size_t)(r0 + 16 + r) * HID_DIM + n0 + 16 + l15] = c11[r];
  }
}

// ---------------------------------------------------------------------------
// RoPE in place on Q and K. One thread per (s,h,d<64) pair.
// ---------------------------------------------------------------------------
__global__ void rope_kernel(float* __restrict__ q, float* __restrict__ k) {
  int idx = blockIdx.x * blockDim.x + threadIdx.x;
  if (idx >= S_LEN * NHEAD * 64) return;
  int d  = idx & 63;
  int hs = idx >> 6;
  int h  = hs & (NHEAD - 1);
  int s  = hs >> 5;
  float inv = __powf(10000.0f, -(float)(2 * d) / 128.0f);
  float ang = (float)s * inv;
  float cs = __cosf(ang), sn = __sinf(ang);
  size_t base = (size_t)s * HID_DIM + h * HDIM + d;
  float q0 = q[base], q1 = q[base + 64];
  q[base]      = q0 * cs - q1 * sn;
  q[base + 64] = q1 * cs + q0 * sn;
  float k0 = k[base], k1 = k[base + 64];
  k[base]      = k0 * cs - k1 * sn;
  k[base + 64] = k1 * cs + k0 * sn;
}

// ---------------------------------------------------------------------------
// qsum[i] = sum over text queries of q[t][i]  (mean-of-dots == dot-with-sum)
// ---------------------------------------------------------------------------
__global__ void qsum_kernel(const float* __restrict__ q, float* __restrict__ qsum) {
  int i = blockIdx.x * blockDim.x + threadIdx.x;
  if (i >= HID_DIM) return;
  float s = 0.0f;
  for (int t = INSTR_START; t < INSTR_END; ++t)
    s += q[(size_t)t * HID_DIM + i];
  qsum[i] = s;
}

// ---------------------------------------------------------------------------
// importance[v] = (scale / (H*T)) * qsum . k[VIS_START+v]
// ---------------------------------------------------------------------------
__global__ void importance_kernel(const float* __restrict__ k,
                                  const float* __restrict__ qsum,
                                  float* __restrict__ imp) {
  int v = blockIdx.x * blockDim.x + threadIdx.x;
  if (v >= NVIS) return;
  const float* krow = k + (size_t)(VIS_START + v) * HID_DIM;
  float acc = 0.0f;
  for (int i = 0; i < HID_DIM; i += 4) {
    float4 kv = *(const float4*)(krow + i);
    float4 qv = *(const float4*)(qsum + i);
    acc += kv.x * qv.x + kv.y * qv.y + kv.z * qv.z + kv.w * qv.w;
  }
  imp[v] = acc * (ATTN_SCALE / (float)(NHEAD * (INSTR_END - INSTR_START)));
}

// ---------------------------------------------------------------------------
// Top-256 selection (lax.top_k tie semantics: ties -> lower index) and
// sorted kept index list: [0] ++ kept_vis_abs(asc) ++ [513..639]
// ---------------------------------------------------------------------------
__global__ __launch_bounds__(512)
void select_kernel(const float* __restrict__ imp, int* __restrict__ kept_all) {
  __shared__ float simp[NVIS];
  __shared__ int   skeep[NVIS];
  int v = threadIdx.x;
  simp[v] = imp[v];
  __syncthreads();
  float mv = simp[v];
  int rank = 0;
  for (int u = 0; u < NVIS; ++u) {
    float uu = simp[u];
    rank += (uu > mv) || (uu == mv && u < v);
  }
  int keep = (rank < NKEEP) ? 1 : 0;
  skeep[v] = keep;
  __syncthreads();
  if (keep) {
    int pos = 0;
    for (int u = 0; u < v; ++u) pos += skeep[u];
    kept_all[1 + pos] = VIS_START + v;
  }
  if (v == 0) kept_all[0] = 0;
  if (v < S_LEN - VIS_END) kept_all[1 + NKEEP + v] = VIS_END + v;
}

// ---------------------------------------------------------------------------
// Pruned causal attention. Block = (head, 16-query tile), 8 waves.
// Phase 1: scores 16x384 via WMMA (D=128 -> 4 chained WMMAs per key tile).
// Phase 2: causal mask + softmax in LDS, probs -> bf16.
// Phase 3: P(16x384) @ V_kept(384x128) via WMMA (12 chained per d-tile).
// ---------------------------------------------------------------------------
__global__ __launch_bounds__(256)
void attn_kernel(const float* __restrict__ q, const float* __restrict__ k,
                 const float* __restrict__ v, const int* __restrict__ kept_all,
                 float* __restrict__ out) {
  __shared__ float  sc[16][NKEPT];   // 24 KB scores / exp
  __shared__ __bf16 sp[16][NKEPT];   // 12 KB probabilities
  __shared__ int    skept[NKEPT];

  const int lane = threadIdx.x & 31;
  const int wave = threadIdx.x >> 5;
  const int half = lane >> 4;
  const int l15  = lane & 15;
  const int h    = blockIdx.x;
  const int q0p  = blockIdx.y * 16;

  for (int i = threadIdx.x; i < NKEPT; i += 256) skept[i] = kept_all[i];
  __syncthreads();

  const float* qrow = q + (size_t)(q0p + l15) * HID_DIM + h * HDIM;

  // -------- phase 1: S = Q Kkept^T * scale --------
  for (int kb = 0; kb < NKEPT; kb += 128) {
    int key = kb + wave * 16 + l15;
    const float* krow = k + (size_t)skept[key] * HID_DIM + h * HDIM;
    v8f c = {};
#pragma unroll
    for (int kt = 0; kt < HDIM; kt += 32) {
      v16bf a = cvtA16(qrow + kt, half * 16);
      v16bf b = cvtB16(krow + kt + half * 16);
      c = wmma_bf16(a, b, c);
    }
#pragma unroll
    for (int r = 0; r < 8; ++r)
      sc[half * 8 + r][kb + wave * 16 + l15] = c[r] * ATTN_SCALE;
  }
  __syncthreads();

  // -------- phase 2: causal mask + softmax per query row --------
  if (threadIdx.x < 16) {
    int m    = threadIdx.x;
    int qpos = q0p + m;
    float mx = -1e30f;
    for (int j = 0; j < NKEPT; ++j) {
      float s = (skept[j] <= qpos) ? sc[m][j] : -1e30f;
      sc[m][j] = s;
      mx = fmaxf(mx, s);
    }
    float sum = 0.0f;
    for (int j = 0; j < NKEPT; ++j) {
      float e = __expf(sc[m][j] - mx);
      sum += e;
      sc[m][j] = e;
    }
    float inv = 1.0f / sum;
    for (int j = 0; j < NKEPT; ++j) sp[m][j] = (__bf16)(sc[m][j] * inv);
  }
  __syncthreads();

  // -------- phase 3: O = P Vkept ; wave owns 16 output dims --------
  {
    int dn = wave * 16 + l15;
    v8f c = {};
    for (int kt = 0; kt < NKEPT; kt += 32) {
      v16bf a;
#pragma unroll
      for (int i = 0; i < 8; ++i) a[i]     = sp[l15][kt + half * 16 + i];
#pragma unroll
      for (int i = 0; i < 8; ++i) a[8 + i] = sp[l15][kt + half * 16 + 8 + i];
      v16bf b;
#pragma unroll
      for (int i = 0; i < 16; ++i)
        b[i] = (__bf16)v[(size_t)skept[kt + half * 16 + i] * HID_DIM + h * HDIM + dn];
      c = wmma_bf16(a, b, c);
    }
    float* orow = out + (size_t)(q0p + half * 8) * HID_DIM + h * HDIM + dn;
#pragma unroll
    for (int r = 0; r < 8; ++r) orow[(size_t)r * HID_DIM] = c[r];
  }
}

// ---------------------------------------------------------------------------
extern "C" void kernel_launch(void* const* d_in, const int* in_sizes, int n_in,
                              void* d_out, int out_size, void* d_ws, size_t ws_size,
                              hipStream_t stream) {
  const float* hs = (const float*)d_in[0];
  const float* qw = (const float*)d_in[1];
  const float* kw = (const float*)d_in[2];
  const float* vw = (const float*)d_in[3];
  const float* ow = (const float*)d_in[4];
  float* out = (float*)d_out;

  const size_t SH = (size_t)S_LEN * HID_DIM;   // 2.62M floats
  float* q    = (float*)d_ws;
  float* k    = q + SH;
  float* v    = k + SH;
  float* ao   = v + SH;          // attention output, pre o-proj
  float* qsum = ao + SH;
  float* imp  = qsum + HID_DIM;
  int*   kept = (int*)(imp + NVIS);

  // fused Q/K/V projections: one launch, blockIdx.z selects weight
  dim3 ggrid(HID_DIM / BLK_N, S_LEN / BLK_M, 3);     // 16 x 20 x 3
  gemm_wmma_kernel<<<ggrid, 256, 0, stream>>>(hs, qw, kw, vw, q, k, v);

  rope_kernel<<<(S_LEN * NHEAD * 64 + 255) / 256, 256, 0, stream>>>(q, k);
  qsum_kernel<<<HID_DIM / 256, 256, 0, stream>>>(q, qsum);
  importance_kernel<<<NVIS / 256, 256, 0, stream>>>(k, qsum, imp);
  select_kernel<<<1, 512, 0, stream>>>(imp, kept);

  attn_kernel<<<dim3(NHEAD, S_LEN / 16), 256, 0, stream>>>(q, k, v, kept, ao);

  // output projection
  dim3 ogrid(HID_DIM / BLK_N, S_LEN / BLK_M, 1);
  gemm_wmma_kernel<<<ogrid, 256, 0, stream>>>(ao, ow, ow, ow, out, out, out);
}
